// TemporalSelfAttention_10101763080305
// MI455X (gfx1250) — matrix operand order repaired and checked by reading, not compiled, not run on hardware
//
#include <hip/hip_runtime.h>
#include <math.h>

#define D_MODEL 1024
#define NHEAD   16
#define DK      64
#define BATCH   4
#define SEQ     2048

typedef __attribute__((ext_vector_type(16))) _Float16 v16h;
typedef __attribute__((ext_vector_type(8)))  _Float16 v8h;
typedef __attribute__((ext_vector_type(8)))  float    v8f;
typedef __attribute__((ext_vector_type(4)))  float    v4f;

// ---------------------------------------------------------------------------
// CDNA5 async copy (global -> LDS), tracked by ASYNCcnt.  GVS addressing:
//   mem = SGPR_base + VGPR_i32_offset;  LDS[vdst_vgpr] = mem   (16B per lane)
// ---------------------------------------------------------------------------
__device__ __forceinline__ void async_lds_b128(unsigned lds_addr, unsigned goff,
                                               const void* base) {
  asm volatile("global_load_async_to_lds_b128 %0, %1, %2 offset:0"
               :: "v"(lds_addr), "v"(goff), "s"(base) : "memory");
}
__device__ __forceinline__ void wait_async_le8() {
  asm volatile("s_wait_asynccnt 0x8" ::: "memory");
}
__device__ __forceinline__ void wait_async_0() {
  asm volatile("s_wait_asynccnt 0x0" ::: "memory");
}
__device__ __forceinline__ unsigned lds_off(const void* p) {
  return (unsigned)(unsigned long long)p;   // low 32 bits of flat addr = LDS offset
}

// ---------------------------------------------------------------------------
// WMMA fragment helpers (f16 16x16x32, wave32).
// A-frag: lane l holds row (l&15); K elems {g*8..g*8+7, g*8+16..g*8+23}, g=l>>4.
// B-frag: lane l holds col (l&15); K elems {g*8..g*8+7, g*8+16..g*8+23}, g=l>>4.
// LDS tiles keep K contiguous (A row-major, B pre-transposed), stride 72 halves.
// ---------------------------------------------------------------------------
__device__ __forceinline__ v16h frag_a(const _Float16* __restrict__ row, int ks, int lane) {
  const int o = ks * 32 + ((lane >> 4) << 3);
  v8h lo = *(const v8h*)(row + o);
  v8h hi = *(const v8h*)(row + o + 16);
  return __builtin_shufflevector(lo, hi, 0,1,2,3,4,5,6,7,8,9,10,11,12,13,14,15);
}
__device__ __forceinline__ v16h frag_b(const _Float16* __restrict__ row, int ks, int lane) {
  const int o = ks * 32 + ((lane >> 4) << 3);
  v8h lo = *(const v8h*)(row + o);
  v8h hi = *(const v8h*)(row + o + 16);
  return __builtin_shufflevector(lo, hi, 0,1,2,3,4,5,6,7,8,9,10,11,12,13,14,15);
}
__device__ __forceinline__ v8f wmma32(v16h a, v16h b, v8f c) {
  return __builtin_amdgcn_wmma_f32_16x16x32_f16(false, a, false, b, (short)0, c, false, false);
}

// ---------------------------------------------------------------------------
// WMMA GEMM, all-f16 operands: C[M,N] = A[M,K] @ Bt[N,K]^T (+bias)(+ELU).
// 128x128x64 tiles, 8 waves x (2x4) WMMA tiles, double-buffered LDS filled by
// async-to-LDS DMA (8 in-flight b128 copies per wave per stage).
// Outputs: optional f32 and/or f16 images.  Batched via grid.z.
// ---------------------------------------------------------------------------
template<int ELU>
__global__ __launch_bounds__(256) void gemm_wmma(
    const _Float16* __restrict__ A, const _Float16* __restrict__ Bt,
    const float* __restrict__ bias,
    float* __restrict__ Cf, _Float16* __restrict__ Ch,
    int M, int N, int K, long long aB, long long bB, long long cB)
{
  __shared__ __align__(16) _Float16 As[2][128][72];
  __shared__ __align__(16) _Float16 Bs[2][128][72];
  A += (long long)blockIdx.z * aB;
  Bt += (long long)blockIdx.z * bB;
  const int m0 = blockIdx.y << 7, n0 = blockIdx.x << 7;
  const int tid = threadIdx.x, lane = tid & 31, wv = tid >> 5;
  const int wm = (wv & 3) << 5, wn = (wv >> 2) << 6;
  const int lr = lane & 15, rbase = (lane >> 4) << 3;

  auto issue = [&](int buf, int k0) {
    const unsigned aA = lds_off(&As[buf][0][0]);
    const unsigned aB2 = lds_off(&Bs[buf][0][0]);
    #pragma unroll
    for (int i = 0; i < 4; i++) {
      const unsigned id = (unsigned)tid + (i << 8);        // 0..1023
      const unsigned r = id >> 3, c16 = (id & 7) << 4;     // 8 x 16B chunks/row
      async_lds_b128(aA  + r * 144 + c16,
                     (unsigned)(((m0 + (int)r) * K + k0) * 2) + c16, A);
      async_lds_b128(aB2 + r * 144 + c16,
                     (unsigned)(((n0 + (int)r) * K + k0) * 2) + c16, Bt);
    }
  };

  v8f acc[2][4] = {};
  issue(0, 0);
  int cur = 0;
  for (int k0 = 0; k0 < K; k0 += 64) {
    if (k0 + 64 < K) { issue(cur ^ 1, k0 + 64); wait_async_le8(); }
    else             { wait_async_0(); }
    __syncthreads();
    #pragma unroll
    for (int ks = 0; ks < 2; ks++) {
      v16h af[2], bf[4];
      #pragma unroll
      for (int mt = 0; mt < 2; mt++) af[mt] = frag_a(&As[cur][wm + mt*16 + lr][0], ks, lane);
      #pragma unroll
      for (int nt = 0; nt < 4; nt++) bf[nt] = frag_b(&Bs[cur][wn + nt*16 + lr][0], ks, lane);
      #pragma unroll
      for (int mt = 0; mt < 2; mt++)
        #pragma unroll
        for (int nt = 0; nt < 4; nt++)
          acc[mt][nt] = wmma32(af[mt], bf[nt], acc[mt][nt]);
    }
    __syncthreads();                       // stage fully consumed before reuse
    cur ^= 1;
  }
  #pragma unroll
  for (int nt = 0; nt < 4; nt++) {
    const int col = n0 + wn + nt*16 + lr;
    const float bc = bias ? bias[col] : 0.f;
    #pragma unroll
    for (int mt = 0; mt < 2; mt++) {
      #pragma unroll
      for (int v = 0; v < 8; v++) {
        const int row = m0 + wm + mt*16 + rbase + v;
        float val = acc[mt][nt][v] + bc;
        if (ELU) val = (val > 0.f) ? val : expm1f(val);
        const long long o = (long long)blockIdx.z * cB + (long long)row * N + col;
        if (Cf) Cf[o] = val;
        if (Ch) Ch[o] = (_Float16)val;
      }
    }
  }
}

// ---------------------------------------------------------------------------
// Pass 1: per (b, h, q-tile) causal softmax stats (row max m, row sum l) over
// Q K^T / 8.  f16 Q/K async-staged; scores via WMMA into LDS; 128 row-threads
// keep online (m, l).
// ---------------------------------------------------------------------------
__global__ __launch_bounds__(256) void attn_stats(
    const _Float16* __restrict__ Qh, const _Float16* __restrict__ Kh,
    float* __restrict__ Mr, float* __restrict__ Lr)
{
  __shared__ __align__(16) _Float16 Qs[128][72];
  __shared__ __align__(16) _Float16 Ks[128][72];
  __shared__ float Sc[128][129];
  const int b = blockIdx.z, h = blockIdx.y, qt = blockIdx.x;
  const int q0 = qt << 7;
  const int tid = threadIdx.x, lane = tid & 31, wv = tid >> 5;
  const int wm = (wv & 3) << 5, wn = (wv >> 2) << 6;
  const int lr = lane & 15, rbase = (lane >> 4) << 3;
  const _Float16* Qb = Qh + ((long long)b * SEQ) * D_MODEL + h * DK;
  const _Float16* Kb = Kh + ((long long)b * SEQ) * D_MODEL + h * DK;
  const unsigned aQ = lds_off(&Qs[0][0]);
  const unsigned aK = lds_off(&Ks[0][0]);

  #pragma unroll
  for (int i = 0; i < 4; i++) {              // stage Q tile once (async)
    const unsigned id = (unsigned)tid + (i << 8);
    const unsigned r = id >> 3, c16 = (id & 7) << 4;
    async_lds_b128(aQ + r * 144 + c16,
                   (unsigned)((q0 + (int)r) * D_MODEL * 2) + c16, Qb);
  }
  float mrun = -INFINITY, lrun = 0.f;
  for (int kt = 0; kt <= qt; kt++) {
    const int k0 = kt << 7;
    __syncthreads();                         // prev stats/WMMA done
    #pragma unroll
    for (int i = 0; i < 4; i++) {            // stage K tile (async)
      const unsigned id = (unsigned)tid + (i << 8);
      const unsigned r = id >> 3, c16 = (id & 7) << 4;
      async_lds_b128(aK + r * 144 + c16,
                     (unsigned)((k0 + (int)r) * D_MODEL * 2) + c16, Kb);
    }
    wait_async_0();
    __syncthreads();
    v8f sacc[2][4] = {};
    #pragma unroll
    for (int ks = 0; ks < 2; ks++) {
      v16h af[2], bf[4];
      #pragma unroll
      for (int mt = 0; mt < 2; mt++) af[mt] = frag_a(&Qs[wm + mt*16 + lr][0], ks, lane);
      #pragma unroll
      for (int nt = 0; nt < 4; nt++) bf[nt] = frag_b(&Ks[wn + nt*16 + lr][0], ks, lane);
      #pragma unroll
      for (int mt = 0; mt < 2; mt++)
        #pragma unroll
        for (int nt = 0; nt < 4; nt++)
          sacc[mt][nt] = wmma32(af[mt], bf[nt], sacc[mt][nt]);
    }
    #pragma unroll
    for (int mt = 0; mt < 2; mt++)
      #pragma unroll
      for (int nt = 0; nt < 4; nt++)
        #pragma unroll
        for (int v = 0; v < 8; v++) {
          const int rl = wm + mt*16 + rbase + v;
          const int cl = wn + nt*16 + lr;
          float s = sacc[mt][nt][v] * 0.125f;
          if (k0 + cl > q0 + rl) s = -INFINITY;   // causal mask
          Sc[rl][cl] = s;
        }
    __syncthreads();
    if (tid < 128) {                         // online (m, l)
      float tmax = -INFINITY;
      for (int j = 0; j < 128; j++) tmax = fmaxf(tmax, Sc[tid][j]);
      const float mnew = fmaxf(mrun, tmax);
      float s = 0.f;
      for (int j = 0; j < 128; j++) s += __expf(Sc[tid][j] - mnew);
      lrun = lrun * __expf(mrun - mnew) + s;
      mrun = mnew;
    }
  }
  if (tid < 128) {
    const long long o = ((long long)(b * NHEAD + h)) * SEQ + q0 + tid;
    Mr[o] = mrun; Lr[o] = lrun;
  }
}

// ---------------------------------------------------------------------------
// Pass 2: per (b, q-tile, k-tile) recompute scores for all heads via WMMA and
// accumulate avg_attn = (1/H) sum_h exp(s-m_h)/l_h.  Writes f32 (d_out) and a
// f16 copy (operand of avg @ V).  Tiles above the diagonal write zeros.
// ---------------------------------------------------------------------------
__global__ __launch_bounds__(256) void attn_avg(
    const _Float16* __restrict__ Qh, const _Float16* __restrict__ Kh,
    const float* __restrict__ Mr, const float* __restrict__ Lr,
    float* __restrict__ avg, _Float16* __restrict__ avgh)
{
  const int b = blockIdx.z, qt = blockIdx.y, kt = blockIdx.x;
  const int q0 = qt << 7, k0 = kt << 7;
  const int tid = threadIdx.x;
  float*     outF = avg  + ((long long)b * SEQ + q0) * SEQ + k0;
  _Float16*  outH = avgh + ((long long)b * SEQ + q0) * SEQ + k0;
  if (kt > qt) {                             // fully masked -> zeros
    v4f zf = {0.f, 0.f, 0.f, 0.f};
    v8h zh = {};
    #pragma unroll
    for (int i = 0; i < 16; i++) {
      const int idx = tid + (i << 8);
      const int r = idx >> 5, c = (idx & 31) << 2;
      *(v4f*)(outF + (long long)r * SEQ + c) = zf;
    }
    #pragma unroll
    for (int i = 0; i < 8; i++) {
      const int idx = tid + (i << 8);
      const int r = idx >> 4, c = (idx & 15) << 3;
      *(v8h*)(outH + (long long)r * SEQ + c) = zh;
    }
    return;
  }
  __shared__ __align__(16) _Float16 Qs[128][72];
  __shared__ __align__(16) _Float16 Ks[128][72];
  __shared__ float mS[NHEAD][128];
  __shared__ float lS[NHEAD][128];
  const int lane = tid & 31, wv = tid >> 5;
  const int wm = (wv & 3) << 5, wn = (wv >> 2) << 6;
  const int lr = lane & 15, rbase = (lane >> 4) << 3;
  const unsigned aQ = lds_off(&Qs[0][0]);
  const unsigned aK = lds_off(&Ks[0][0]);

  #pragma unroll
  for (int i = 0; i < 8; i++) {              // stage per-head row stats
    const int idx = tid + (i << 8);
    const int hh = idx >> 7, r = idx & 127;
    const long long o = ((long long)(b * NHEAD + hh)) * SEQ + q0 + r;
    mS[hh][r] = Mr[o];
    lS[hh][r] = Lr[o];
  }
  float aacc[2][4][8] = {};
  for (int h = 0; h < NHEAD; h++) {
    __syncthreads();
    const _Float16* Qb = Qh + ((long long)b * SEQ) * D_MODEL + h * DK;
    const _Float16* Kb = Kh + ((long long)b * SEQ) * D_MODEL + h * DK;
    #pragma unroll
    for (int i = 0; i < 4; i++) {
      const unsigned id = (unsigned)tid + (i << 8);
      const unsigned r = id >> 3, c16 = (id & 7) << 4;
      async_lds_b128(aQ + r * 144 + c16,
                     (unsigned)((q0 + (int)r) * D_MODEL * 2) + c16, Qb);
      async_lds_b128(aK + r * 144 + c16,
                     (unsigned)((k0 + (int)r) * D_MODEL * 2) + c16, Kb);
    }
    wait_async_0();
    __syncthreads();
    v8f sacc[2][4] = {};
    #pragma unroll
    for (int ks = 0; ks < 2; ks++) {
      v16h af[2], bf[4];
      #pragma unroll
      for (int mt = 0; mt < 2; mt++) af[mt] = frag_a(&Qs[wm + mt*16 + lr][0], ks, lane);
      #pragma unroll
      for (int nt = 0; nt < 4; nt++) bf[nt] = frag_b(&Ks[wn + nt*16 + lr][0], ks, lane);
      #pragma unroll
      for (int mt = 0; mt < 2; mt++)
        #pragma unroll
        for (int nt = 0; nt < 4; nt++)
          sacc[mt][nt] = wmma32(af[mt], bf[nt], sacc[mt][nt]);
    }
    #pragma unroll
    for (int mt = 0; mt < 2; mt++)
      #pragma unroll
      for (int nt = 0; nt < 4; nt++)
        #pragma unroll
        for (int v = 0; v < 8; v++) {
          const int rl = wm + mt*16 + rbase + v;
          const int cl = wn + nt*16 + lr;
          if (k0 + cl <= q0 + rl) {
            const float s = sacc[mt][nt][v] * 0.125f;
            aacc[mt][nt][v] += __expf(s - mS[h][rl]) / lS[h][rl];
          }
        }
  }
  #pragma unroll
  for (int mt = 0; mt < 2; mt++)
    #pragma unroll
    for (int nt = 0; nt < 4; nt++)
      #pragma unroll
      for (int v = 0; v < 8; v++) {
        const int rl = wm + mt*16 + rbase + v;
        const int cl = wn + nt*16 + lr;
        const float a = aacc[mt][nt][v] * (1.f / 16.f);
        outF[(long long)rl * SEQ + cl] = a;
        outH[(long long)rl * SEQ + cl] = (_Float16)a;
      }
}

// ---------------------------------------------------------------------------
// GRN tail: glu = sigmoid(z[:,D:]) * z[:,:D]; out = LN(x + glu)*g + b.
// Optional f32 / f16 outputs.
// ---------------------------------------------------------------------------
__global__ __launch_bounds__(256) void grn_finish(
    const float* __restrict__ x, const float* __restrict__ z,
    const float* __restrict__ g, const float* __restrict__ bta,
    float* __restrict__ outf, _Float16* __restrict__ outh)
{
  const long long row = blockIdx.x;
  const float* zr = z + row * (2 * D_MODEL);
  const float* xr = x + row * D_MODEL;
  const int tid = threadIdx.x;
  float t[4], sum = 0.f, ss = 0.f;
  #pragma unroll
  for (int j = 0; j < 4; j++) {
    const int i = tid + (j << 8);
    const float v = zr[i], gt = zr[i + D_MODEL];
    const float glu = v / (1.f + __expf(-gt));
    const float tv = xr[i] + glu;
    t[j] = tv; sum += tv; ss += tv * tv;
  }
  __shared__ float red[16];
  __shared__ float stats[2];
  #pragma unroll
  for (int off = 16; off > 0; off >>= 1) {
    sum += __shfl_down(sum, off, 32);
    ss  += __shfl_down(ss,  off, 32);
  }
  if ((tid & 31) == 0) { red[tid >> 5] = sum; red[8 + (tid >> 5)] = ss; }
  __syncthreads();
  if (tid == 0) {
    float s = 0.f, q = 0.f;
    for (int w = 0; w < 8; w++) { s += red[w]; q += red[8 + w]; }
    stats[0] = s * (1.f / (float)D_MODEL);
    stats[1] = q * (1.f / (float)D_MODEL);
  }
  __syncthreads();
  const float mean = stats[0];
  const float rstd = rsqrtf(stats[1] - mean * mean + 1e-5f);
  #pragma unroll
  for (int j = 0; j < 4; j++) {
    const int i = tid + (j << 8);
    const float o = (t[j] - mean) * rstd * g[i] + bta[i];
    if (outf) outf[row * D_MODEL + i] = o;
    if (outh) outh[row * D_MODEL + i] = (_Float16)o;
  }
}

// ---------------------------------------------------------------------------
// y = x + sinusoidal PE (f32 residual copy + f16 GEMM-operand copy).
// ---------------------------------------------------------------------------
__global__ __launch_bounds__(256) void pe_add(const float* __restrict__ x,
                                              float* __restrict__ y,
                                              _Float16* __restrict__ yh)
{
  const long long idx = (long long)blockIdx.x * 256 + threadIdx.x;
  const int d = (int)(idx & (D_MODEL - 1));
  const int s = (int)((idx >> 10) & (SEQ - 1));
  const float dv = expf((float)(d & ~1) * (-9.210340371976184f / (float)D_MODEL));
  const float a = (float)s * dv;
  const float pe = (d & 1) ? cosf(a) : sinf(a);
  const float v = x[idx] + pe;
  y[idx] = v;
  yh[idx] = (_Float16)v;
}

// ---------------------------------------------------------------------------
// Wt[n][k] = (f16) W[k][n]  (32x32 LDS tiles; batched via grid.z).
// ---------------------------------------------------------------------------
__global__ __launch_bounds__(256) void transpose_cvt(
    const float* __restrict__ W, _Float16* __restrict__ Wt,
    int K, int N, long long wB, long long tB)
{
  __shared__ float tile[32][33];
  W  += (long long)blockIdx.z * wB;
  Wt += (long long)blockIdx.z * tB;
  const int kb = blockIdx.y << 5, nb = blockIdx.x << 5;
  const int tx = threadIdx.x & 31, ty = threadIdx.x >> 5;
  #pragma unroll
  for (int i = 0; i < 32; i += 8)
    tile[ty + i][tx] = W[(long long)(kb + ty + i) * N + nb + tx];
  __syncthreads();
  #pragma unroll
  for (int i = 0; i < 32; i += 8)
    Wt[(long long)(nb + ty + i) * K + kb + tx] = (_Float16)tile[tx][ty + i];
}

// ---------------------------------------------------------------------------
extern "C" void kernel_launch(void* const* d_in, const int* in_sizes, int n_in,
                              void* d_out, int out_size, void* d_ws, size_t ws_size,
                              hipStream_t stream)
{
  const float* x   = (const float*)d_in[0];
  const float* pw1 = (const float*)d_in[1];  const float* pb1 = (const float*)d_in[2];
  const float* pw2 = (const float*)d_in[3];  const float* pb2 = (const float*)d_in[4];
  const float* pwg = (const float*)d_in[5];  const float* pbg = (const float*)d_in[6];
  const float* plg = (const float*)d_in[7];  const float* plb = (const float*)d_in[8];
  const float* sw1 = (const float*)d_in[9];  const float* sb1 = (const float*)d_in[10];
  const float* sw2 = (const float*)d_in[11]; const float* sb2 = (const float*)d_in[12];
  const float* swg = (const float*)d_in[13]; const float* sbg = (const float*)d_in[14];
  const float* slg = (const float*)d_in[15]; const float* slb = (const float*)d_in[16];
  const float* wq  = (const float*)d_in[17]; const float* bq  = (const float*)d_in[18];
  const float* wk  = (const float*)d_in[19]; const float* bk  = (const float*)d_in[20];
  const float* wv  = (const float*)d_in[21]; const float* bv  = (const float*)d_in[22];
  const float* wo  = (const float*)d_in[23]; const float* bo  = (const float*)d_in[24];

  float* outp = (float*)d_out;
  const long long MT  = (long long)BATCH * SEQ * D_MODEL;   // 8,388,608
  const long long BHS = (long long)BATCH * NHEAD * SEQ;
  const long long DD  = (long long)D_MODEL * D_MODEL;       // 1,048,576
  float* avgp = outp + MT;

  float* ws    = (float*)d_ws;
  float* F_res = ws;                 // x+pe (residual), later proj f32
  float* F_V   = F_res + MT;         // V f32 (pre-transpose)
  float* F_Z   = F_V + MT;           // GLU pre-activation (2*MT)
  float* mbuf  = F_Z + 2 * MT;
  float* lbuf  = mbuf + BHS;
  _Float16* Hx1  = (_Float16*)(lbuf + BHS); // x+pe f16 / Q / proj f16
  _Float16* H1   = Hx1 + MT;                // h1 / K / post-h1
  _Float16* H2   = H1 + MT;                 // h2 / t=avg@V / post-h2
  _Float16* Hxln = H2 + MT;                 // pre-GRN output f16
  _Float16* HVt  = Hxln + MT;               // V transposed f16 (B,D,S)
  _Float16* Havg = HVt + MT;                // avg_attn f16 (2*MT halves)
  _Float16* Wts  = Havg + 2 * MT;           // transposed weights arena
  _Float16* Tpw1 = Wts;            _Float16* Tpw2 = Tpw1 + DD;
  _Float16* Tpwg = Tpw2 + DD;      _Float16* Twq  = Tpwg + 2 * DD;
  _Float16* Twk  = Twq + DD;       _Float16* Twv  = Twk + DD;
  _Float16* Two  = Twv + DD;       _Float16* Tsw1 = Two + DD;
  _Float16* Tsw2 = Tsw1 + DD;      _Float16* Tswg = Tsw2 + DD;

  const dim3 blk(256);
  const dim3 tg1(32, 32, 1), tg2(64, 32, 1);
  const dim3 gN1(1024 / 128, 8192 / 128, 1);
  const dim3 gN2(2048 / 128, 8192 / 128, 1);

  // 0. one-time weight transpose + f16 convert (Bt layout: N x K)
  transpose_cvt<<<tg1, blk, 0, stream>>>(pw1, Tpw1, 1024, 1024, 0, 0);
  transpose_cvt<<<tg1, blk, 0, stream>>>(pw2, Tpw2, 1024, 1024, 0, 0);
  transpose_cvt<<<tg2, blk, 0, stream>>>(pwg, Tpwg, 1024, 2048, 0, 0);
  transpose_cvt<<<tg1, blk, 0, stream>>>(wq,  Twq,  1024, 1024, 0, 0);
  transpose_cvt<<<tg1, blk, 0, stream>>>(wk,  Twk,  1024, 1024, 0, 0);
  transpose_cvt<<<tg1, blk, 0, stream>>>(wv,  Twv,  1024, 1024, 0, 0);
  transpose_cvt<<<tg1, blk, 0, stream>>>(wo,  Two,  1024, 1024, 0, 0);
  transpose_cvt<<<tg1, blk, 0, stream>>>(sw1, Tsw1, 1024, 1024, 0, 0);
  transpose_cvt<<<tg1, blk, 0, stream>>>(sw2, Tsw2, 1024, 1024, 0, 0);
  transpose_cvt<<<tg2, blk, 0, stream>>>(swg, Tswg, 1024, 2048, 0, 0);
  // 1. positional encoding
  pe_add<<<(unsigned)(MT / 256), blk, 0, stream>>>(x, F_res, Hx1);
  // 2. pre-attention GRN
  gemm_wmma<1><<<gN1, blk, 0, stream>>>(Hx1, Tpw1, pb1, nullptr, H1, 8192, 1024, 1024, 0, 0, 0);
  gemm_wmma<0><<<gN1, blk, 0, stream>>>(H1, Tpw2, pb2, nullptr, H2, 8192, 1024, 1024, 0, 0, 0);
  gemm_wmma<0><<<gN2, blk, 0, stream>>>(H2, Tpwg, pbg, F_Z, nullptr, 8192, 2048, 1024, 0, 0, 0);
  grn_finish<<<8192, blk, 0, stream>>>(F_res, F_Z, plg, plb, nullptr, Hxln);
  // 3. Q / K / V projections (Q,K f16-only; V f32 then batched transpose)
  gemm_wmma<0><<<gN1, blk, 0, stream>>>(Hxln, Twq, bq, nullptr, Hx1, 8192, 1024, 1024, 0, 0, 0);
  gemm_wmma<0><<<gN1, blk, 0, stream>>>(Hxln, Twk, bk, nullptr, H1, 8192, 1024, 1024, 0, 0, 0);
  gemm_wmma<0><<<gN1, blk, 0, stream>>>(Hxln, Twv, bv, F_V, nullptr, 8192, 1024, 1024, 0, 0, 0);
  transpose_cvt<<<dim3(32, 64, BATCH), blk, 0, stream>>>(
      F_V, HVt, 2048, 1024, (long long)SEQ * D_MODEL, (long long)SEQ * D_MODEL);
  // 4. causal softmax stats + head-averaged attention matrix
  attn_stats<<<dim3(SEQ / 128, NHEAD, BATCH), blk, 0, stream>>>(Hx1, H1, mbuf, lbuf);
  attn_avg  <<<dim3(SEQ / 128, SEQ / 128, BATCH), blk, 0, stream>>>(Hx1, H1, mbuf, lbuf, avgp, Havg);
  // 5. attn_out = avg_attn @ V (batched), then output projection
  gemm_wmma<0><<<dim3(1024 / 128, 2048 / 128, BATCH), blk, 0, stream>>>(
      Havg, HVt, nullptr, nullptr, H2, 2048, 1024, 2048,
      (long long)SEQ * SEQ, (long long)SEQ * D_MODEL, (long long)SEQ * D_MODEL);
  gemm_wmma<0><<<gN1, blk, 0, stream>>>(H2, Two, bo, F_res, Hx1, 8192, 1024, 1024, 0, 0, 0);
  // 6. post-attention GRN -> d_out
  gemm_wmma<1><<<gN1, blk, 0, stream>>>(Hx1, Tsw1, sb1, nullptr, H1, 8192, 1024, 1024, 0, 0, 0);
  gemm_wmma<0><<<gN1, blk, 0, stream>>>(H1, Tsw2, sb2, nullptr, H2, 8192, 1024, 1024, 0, 0, 0);
  gemm_wmma<0><<<gN2, blk, 0, stream>>>(H2, Tswg, sbg, F_Z, nullptr, 8192, 2048, 1024, 0, 0, 0);
  grn_finish<<<8192, blk, 0, stream>>>(F_res, F_Z, slg, slb, outp, nullptr);
}
